// GEN_81192061764274
// MI455X (gfx1250) — compile-verified, matrix operand order checked by reading, not run on hardware
//
#include <hip/hip_runtime.h>
#include <math.h>

// ---------------- problem constants ----------------
#define BQ 16
#define TQ 12
#define NQ 400
#define MQ 295
#define DQ 100
#define FTQ 78
#define NROWS (BQ*TQ*NQ)      // 76800 flattened (b,t,n)
#define BNQ (BQ*NQ)           // 6400
#define NBT (BQ*TQ)           // 192

// padded dims (K paddings are zero-filled so inner loops are branch-free)
#define DP 128                // D padded (K over channels)
#define NP 416                // N padded (K over graph nodes)
#define MP 320                // M padded (K over time features)
#define FP 96                 // FT padded (K over conv positions)
#define KWP 128               // per-tap K stride for conv im2col

typedef __bf16 bf16x16 __attribute__((ext_vector_type(16)));
typedef float  floatx8 __attribute__((ext_vector_type(8)));
typedef unsigned short ushortx8 __attribute__((ext_vector_type(8)));

union Frag { ushortx8 h[2]; bf16x16 bf; };

__device__ __forceinline__ __bf16 u16_bf(unsigned short u) {
  return __builtin_bit_cast(__bf16, u);
}
__device__ __forceinline__ unsigned short f2bf(float f) {
  unsigned int u = __builtin_bit_cast(unsigned int, f);
  u += 0x7FFFu + ((u >> 16) & 1u);   // RNE
  return (unsigned short)(u >> 16);
}
__device__ __forceinline__ floatx8 wmma_bf16(bf16x16 a, bf16x16 b, floatx8 c) {
  return __builtin_amdgcn_wmma_f32_16x16x32_bf16(false, a, false, b, (short)0, c,
                                                 false, false);
}

// Fragment (A row-major or B^T): lane owns a row; its 16 bf16 K-values are two
// contiguous 8-element runs at kbase and kbase+16 (ISA 16-bit A/B VGPR layout).
// All addresses are 16B-aligned by construction (pitches multiple of 16B).
__device__ __forceinline__ bf16x16 ld_frag(const unsigned short* __restrict__ row,
                                           int kbase) {
  Frag f;
  f.h[0] = *(const ushortx8*)(row + kbase);
  f.h[1] = *(const ushortx8*)(row + kbase + 16);
  return f.bf;
}

// ======================= small / elementwise kernels =======================

__global__ __launch_bounds__(256) void k_cvt_sq(const float* __restrict__ s,
                                                unsigned short* __restrict__ d, int nmats) {
  // src: [nmats][100k][100c] row-major -> dst: [nmats][DP c][DP k] (B^T, zero-padded)
  int i = blockIdx.x * blockDim.x + threadIdx.x;
  if (i >= nmats * DQ * DQ) return;
  int mat = i / (DQ * DQ);
  int rem = i - mat * DQ * DQ;
  int k = rem / DQ, c = rem - (rem / DQ) * DQ;
  d[((size_t)mat * DP + c) * DP + k] = f2bf(s[i]);
}

__global__ __launch_bounds__(256) void k_cvt_conv(const float* __restrict__ s,
                                                  unsigned short* __restrict__ d, int w) {
  // src: [w][100 din][100 c] -> dst: [128 c][w*KWP] with K = j*KWP+din
  int i = blockIdx.x * blockDim.x + threadIdx.x;
  if (i >= w * DQ * DQ) return;
  int j = i / (DQ * DQ);
  int rem = i - j * DQ * DQ;
  int din = rem / DQ, c = rem - din * DQ;
  d[(size_t)c * (w * KWP) + j * KWP + din] = f2bf(s[i]);
}

__global__ __launch_bounds__(256) void k_cvt_time(const float* __restrict__ s,
                                                  unsigned short* __restrict__ d) {
  int i = blockIdx.x * blockDim.x + threadIdx.x;
  if (i >= BQ * TQ * MQ) return;
  int bt = i / MQ, m = i - bt * MQ;
  d[(size_t)bt * MP + m] = f2bf(s[i]);
}

__global__ __launch_bounds__(256) void k_cvt_pmat(const float* __restrict__ s,
                                                  unsigned short* __restrict__ d) {
  int i = blockIdx.x * blockDim.x + threadIdx.x;
  if (i >= NQ * NQ) return;
  int r = i / NQ, c = i - r * NQ;
  d[(size_t)r * NP + c] = f2bf(s[i]);
}

__global__ __launch_bounds__(256) void k_fsum(const int* __restrict__ F,
                                              float* __restrict__ rowsum,
                                              float* __restrict__ colsum) {
  int i = blockIdx.x * blockDim.x + threadIdx.x;
  if (i >= NQ) return;
  long rs = 0, cs = 0;
  for (int j = 0; j < NQ; ++j) { rs += F[i * NQ + j]; cs += F[j * NQ + i]; }
  rowsum[i] = (float)rs;
  colsum[i] = (float)cs;
}

__global__ void k_total(const float* __restrict__ rowsum, float* __restrict__ stot) {
  if (blockIdx.x == 0 && threadIdx.x == 0) {
    float s = 0.f;
    for (int i = 0; i < NQ; ++i) s += rowsum[i];
    stot[0] = s;
  }
}

// P[i][j] = max(log(F*s/(colsum[i]*rowsum[j])),0) into padded [400][NP]
__global__ __launch_bounds__(256) void k_pmi(const int* __restrict__ F,
                                             const float* __restrict__ rowsum,
                                             const float* __restrict__ colsum,
                                             const float* __restrict__ stot,
                                             unsigned short* __restrict__ P) {
  int idx = blockIdx.x * blockDim.x + threadIdx.x;
  if (idx >= NQ * NQ) return;
  int i = idx / NQ, j = idx - i * NQ;
  float s = stot[0];
  float v = logf(((float)F[idx] * s) / (colsum[i] * rowsum[j]));
  P[(size_t)i * NP + j] = f2bf(fmaxf(v, 0.f));
}

// zero pad rows 78..95 of comb [6400][FP][100]
__global__ __launch_bounds__(256) void k_pad_comb(unsigned short* __restrict__ comb) {
  int i = blockIdx.x * blockDim.x + threadIdx.x;
  if (i >= BNQ * 18 * DQ) return;
  int bn = i / (18 * DQ);
  int rem = i - bn * 18 * DQ;
  int k = FTQ + rem / DQ, c = rem - (rem / DQ) * DQ;
  comb[((size_t)bn * FP + k) * DQ + c] = 0;
}

// xb = relu(relu(feat@d1+b1)@d2+b2); writes xb_row[76800][DP] (pad cols zero),
// xb_t[NBT][DP][NP] (B^T for P@xb), and comb rows t=0..11.
__global__ __launch_bounds__(256) void k_xb(const float* __restrict__ feat,
                                            const float* __restrict__ d1w, const float* __restrict__ d1b,
                                            const float* __restrict__ d2w, const float* __restrict__ d2b,
                                            unsigned short* __restrict__ xbr,
                                            unsigned short* __restrict__ xbt,
                                            unsigned short* __restrict__ comb) {
  int row = blockIdx.x * blockDim.x + threadIdx.x;
  if (row >= NROWS) return;
  float f0 = feat[row * 2 + 0], f1 = feat[row * 2 + 1];
  float h[10];
#pragma unroll
  for (int j = 0; j < 10; ++j)
    h[j] = fmaxf(f0 * d1w[j] + f1 * d1w[10 + j] + d1b[j], 0.f);
  int bt = row / NQ;
  int n = row - bt * NQ;
  int b = bt / TQ, t = bt - b * TQ;
  int bn = b * NQ + n;
  size_t xrB = (size_t)row * DP;
  size_t xtB = (size_t)bt * DP * NP + n;
  size_t cbB = ((size_t)bn * FP + t) * DQ;
  for (int d = 0; d < DP; ++d) {
    unsigned short u = 0;
    if (d < DQ) {
      float v = d2b[d];
#pragma unroll
      for (int j = 0; j < 10; ++j) v += h[j] * d2w[j * DQ + d];
      u = f2bf(fmaxf(v, 0.f));
      xbt[xtB + (size_t)d * NP] = u;   // coalesced across n-adjacent threads
      comb[cbB + d] = u;
    }
    xbr[xrB + d] = u;
  }
}

// ======================= WMMA kernels =======================

// t1[p][bt] = P_p @ xb[bt]: A=P3 [400][NP], B=xb_t [DP][NP] (B^T), K=NP.
// Register-blocked: one wave owns a 16x64 C strip (4 accumulators); the A
// fragment is loaded once per K chunk and reused for 4 WMMAs. The cg=1 strip
// covers cols 64..127 (100..127 zero pad), so t1's K-pad is written for free.
__global__ __launch_bounds__(256) void k_gcn_t1(const unsigned short* __restrict__ P3,
                                                const unsigned short* __restrict__ xbt,
                                                unsigned short* __restrict__ t1) {
  const int CG = 2, RT = NQ / 16;
  int wid = (blockIdx.x * blockDim.x + threadIdx.x) >> 5;
  if (wid >= 3 * NBT * RT * CG) return;
  int cg = wid % CG; int tmp = wid / CG;
  int rt = tmp % RT; tmp /= RT;
  int bt = tmp % NBT; int p = tmp / NBT;

  int lane = threadIdx.x & 31;
  int kh = (lane >> 4) * 8;
  const unsigned short* Arow = P3 + ((size_t)p * NQ + rt * 16 + (lane & 15)) * NP;
  const unsigned short* Brow = xbt + ((size_t)bt * DP + cg * 64 + (lane & 15)) * NP;

  floatx8 z8 = {0, 0, 0, 0, 0, 0, 0, 0};
  floatx8 acc0 = z8, acc1 = z8, acc2 = z8, acc3 = z8;
#pragma unroll 1
  for (int k0 = 0; k0 < NP; k0 += 32) {   // 13 chunks, pads zero
    bf16x16 a = ld_frag(Arow, k0 + kh);
    acc0 = wmma_bf16(a, ld_frag(Brow,               k0 + kh), acc0);
    acc1 = wmma_bf16(a, ld_frag(Brow + 16 * NP,     k0 + kh), acc1);
    acc2 = wmma_bf16(a, ld_frag(Brow + 32 * NP,     k0 + kh), acc2);
    acc3 = wmma_bf16(a, ld_frag(Brow + 48 * NP,     k0 + kh), acc3);
  }
  unsigned short* C = t1 + ((size_t)p * NROWS + (size_t)bt * NQ) * DP;
  int c0 = cg * 64 + (lane & 15);
  int rb = rt * 16 + (lane >> 4) * 8;
#pragma unroll
  for (int i = 0; i < 8; ++i) {
    size_t idx = (size_t)(rb + i) * DP + c0;
    C[idx]      = f2bf(acc0[i]);
    C[idx + 16] = f2bf(acc1[i]);
    C[idx + 32] = f2bf(acc2[i]);
    C[idx + 48] = f2bf(acc3[i]);
  }
}

// gcnx = sum_p relu(xb@Wp0) + relu(t1_p@Wp1); CT=8 writes all DP cols.
// Phase 1 shares one A fragment (xb) across the three Wp0 products.
__global__ __launch_bounds__(256) void k_gcnx(const unsigned short* __restrict__ xbr,
                                              const unsigned short* __restrict__ t1,
                                              const unsigned short* __restrict__ gW, // [6][DP][DP] B^T
                                              float* __restrict__ gf,
                                              unsigned short* __restrict__ gb) {
  const int CT = 8, RT = NROWS / 16;
  int wid = (blockIdx.x * blockDim.x + threadIdx.x) >> 5;
  if (wid >= RT * CT) return;
  int ct = wid % CT;
  int rt = wid / CT;
  int lane = threadIdx.x & 31;
  int kh = (lane >> 4) * 8;
  int r = rt * 16 + (lane & 15);
  int c = ct * 16 + (lane & 15);

  floatx8 z8 = {0, 0, 0, 0, 0, 0, 0, 0};
  floatx8 sum = z8;
  {  // term 0: relu(xb @ Wp0), shared A across p
    const unsigned short* Arow = xbr + (size_t)r * DP;
    floatx8 a0 = z8, a1 = z8, a2 = z8;
#pragma unroll
    for (int k0 = 0; k0 < DP; k0 += 32) {
      bf16x16 a = ld_frag(Arow, k0 + kh);
      a0 = wmma_bf16(a, ld_frag(gW + ((size_t)0 * DP + c) * DP, k0 + kh), a0);
      a1 = wmma_bf16(a, ld_frag(gW + ((size_t)2 * DP + c) * DP, k0 + kh), a1);
      a2 = wmma_bf16(a, ld_frag(gW + ((size_t)4 * DP + c) * DP, k0 + kh), a2);
    }
#pragma unroll
    for (int i = 0; i < 8; ++i)
      sum[i] += fmaxf(a0[i], 0.f) + fmaxf(a1[i], 0.f) + fmaxf(a2[i], 0.f);
  }
#pragma unroll 1
  for (int p = 0; p < 3; ++p) {  // term 1: relu(t1_p @ Wp1)
    const unsigned short* Arow = t1 + (size_t)p * NROWS * DP + (size_t)r * DP;
    const unsigned short* Brow = gW + ((size_t)(2 * p + 1) * DP + c) * DP;
    floatx8 acc = z8;
#pragma unroll
    for (int k0 = 0; k0 < DP; k0 += 32) {
      bf16x16 a = ld_frag(Arow, k0 + kh);
      bf16x16 b = ld_frag(Brow, k0 + kh);
      acc = wmma_bf16(a, b, acc);
    }
#pragma unroll
    for (int i = 0; i < 8; ++i) sum[i] += fmaxf(acc[i], 0.f);
  }
  int rb = rt * 16 + (lane >> 4) * 8;
#pragma unroll
  for (int i = 0; i < 8; ++i) {
    size_t idx = (size_t)(rb + i) * DP + c;
    gf[idx] = sum[i];
    gb[idx] = f2bf(sum[i]);
  }
}

// multi-kernel temporal conv as im2col GEMM with per-tap K padding (K=w*KWP);
// register-blocked 16x64 strips like k_gcn_t1.
__constant__ int c_cumTiles[12] = {0, 4400, 8400, 12000, 15200, 18000, 20400,
                                   22400, 24000, 25200, 26000, 26400};
__constant__ int c_startpos[11] = {12, 23, 33, 42, 50, 57, 63, 68, 72, 75, 77};
__constant__ int c_wtOff[11] = {0, 2*128*128, 5*128*128, 9*128*128, 14*128*128,
                                20*128*128, 27*128*128, 35*128*128, 44*128*128,
                                54*128*128, 65*128*128};

__global__ __launch_bounds__(256) void k_conv(const unsigned short* __restrict__ xbr,
                                              const unsigned short* __restrict__ cW,
                                              const float* __restrict__ conv_b,
                                              unsigned short* __restrict__ comb) {
  const int CG = 2;
  int wid = (blockIdx.x * blockDim.x + threadIdx.x) >> 5;
  if (wid >= 26400 * CG) return;
  int cg = wid % CG;
  int rw = wid / CG;
  int k = 0;
  while (rw >= c_cumTiles[k + 1]) ++k;    // wave-uniform
  int L = 11 - k, w = k + 2;
  int Kdim = w * KWP;
  int row0 = (rw - c_cumTiles[k]) * 16;

  int lane = threadIdx.x & 31;
  int kh = (lane >> 4) * 8;
  int rA = row0 + (lane & 15);
  int bnA = rA / L;
  int posA = rA - bnA * L;
  int bA = bnA / NQ;
  int nA = bnA - bA * NQ;
  const unsigned short* Brow = cW + c_wtOff[k] + (size_t)(cg * 64 + (lane & 15)) * Kdim;

  floatx8 z8 = {0, 0, 0, 0, 0, 0, 0, 0};
  floatx8 accs[4] = {z8, z8, z8, z8};
#pragma unroll 1
  for (int k0 = 0; k0 < Kdim; k0 += 32) {
    int rs0 = k0 + kh, rs1 = rs0 + 16;
    int j0 = rs0 >> 7, d0 = rs0 & 127;
    int j1 = rs1 >> 7, d1 = rs1 & 127;
    Frag fa;
    fa.h[0] = *(const ushortx8*)(xbr + (size_t)((bA * TQ + posA + j0) * NQ + nA) * DP + d0);
    fa.h[1] = *(const ushortx8*)(xbr + (size_t)((bA * TQ + posA + j1) * NQ + nA) * DP + d1);
#pragma unroll
    for (int j = 0; j < 4; ++j)
      accs[j] = wmma_bf16(fa.bf, ld_frag(Brow + (size_t)j * 16 * Kdim, k0 + kh), accs[j]);
  }
  int rbase = (lane >> 4) * 8;
#pragma unroll
  for (int i = 0; i < 8; ++i) {
    int r = row0 + rbase + i;
    int bn = r / L;
    int pos = r - bn * L;
    size_t base = ((size_t)bn * FP + c_startpos[k] + pos) * DQ;
#pragma unroll
    for (int j = 0; j < 4; ++j) {
      int c = cg * 64 + j * 16 + (lane & 15);
      if (c < DQ) {
        float v = fmaxf(accs[j][i] + conv_b[k * DQ + c], 0.f);
        comb[base + c] = f2bf(v);
      }
    }
  }
}

// att: time[16b x M] @ tb_w[t,n] (M x 78) -> stored [bn][16 t][FP f]
__global__ __launch_bounds__(256) void k_att(const unsigned short* __restrict__ timep,
                                             const float* __restrict__ tbw,
                                             const float* __restrict__ tbb,
                                             unsigned short* __restrict__ att) {
  const int FTT = 5;
  int wid = (blockIdx.x * blockDim.x + threadIdx.x) >> 5;
  if (wid >= TQ * NQ * FTT) return;
  int ft = wid % FTT; int tmp = wid / FTT;
  int n = tmp % NQ;
  int t = tmp / NQ;

  int lane = threadIdx.x & 31;
  int kh = (lane >> 4) * 8;
  const unsigned short* Arow = timep + (size_t)((lane & 15) * TQ + t) * MP;
  int cc = ft * 16 + (lane & 15);
  bool cok = cc < FTQ;
  int ccl = cok ? cc : (FTQ - 1);
  const float* Bp = tbw + ((size_t)t * NQ + n) * MQ * FTQ;

  floatx8 acc = {0, 0, 0, 0, 0, 0, 0, 0};
#pragma unroll 1
  for (int k0 = 0; k0 < 288; k0 += 32) {  // full chunks: m <= 287 < 295
    __builtin_prefetch(Bp + (size_t)(k0 + 32) * FTQ + ccl, 0, 1);
    bf16x16 a = ld_frag(Arow, k0 + kh);
    bf16x16 b;
#pragma unroll
    for (int i = 0; i < 16; ++i) {
      int kk = (i < 8) ? (kh + i) : (16 + kh + (i - 8));
      unsigned short u = cok ? f2bf(Bp[(size_t)(k0 + kk) * FTQ + ccl]) : (unsigned short)0;
      b[i] = u16_bf(u);
    }
    acc = wmma_bf16(a, b, acc);
  }
  { // remainder chunk k0=288, per-element clamp m<295
    bf16x16 a = ld_frag(Arow, 288 + kh);
    bf16x16 b;
#pragma unroll
    for (int i = 0; i < 16; ++i) {
      int kk = (i < 8) ? (kh + i) : (16 + kh + (i - 8));
      int m = 288 + kk;
      bool mok = m < MQ;
      int mcl = mok ? m : (MQ - 1);
      unsigned short u = (cok && mok) ? f2bf(Bp[(size_t)mcl * FTQ + ccl]) : (unsigned short)0;
      b[i] = u16_bf(u);
    }
    acc = wmma_bf16(a, b, acc);
  }
  int rbase = (lane >> 4) * 8;
  if (cok) {
    float bias = tbb[((size_t)t * NQ + n) * FTQ + cc];
#pragma unroll
    for (int i = 0; i < 8; ++i) {
      int b_ = rbase + i;   // batch row, all 16 valid
      att[(((size_t)b_ * NQ + n) * 16 + t) * FP + cc] = f2bf(acc[i] + bias);
    }
  }
}

// re1 per (b,n): att[16t x FP] @ comb[FP x 100]; CT=8, outputs padded to DP
__global__ __launch_bounds__(256) void k_re1(const unsigned short* __restrict__ att,
                                             const unsigned short* __restrict__ comb,
                                             float* __restrict__ rf,
                                             unsigned short* __restrict__ rb) {
  const int CT = 8;
  int wid = (blockIdx.x * blockDim.x + threadIdx.x) >> 5;
  if (wid >= BQ * NQ * CT) return;
  int ct = wid % CT; int tmp = wid / CT;
  int n = tmp % NQ;
  int b = tmp / NQ;
  int bn = b * NQ + n;

  int lane = threadIdx.x & 31;
  int kh = (lane >> 4) * 8;
  const unsigned short* Arow = att + ((size_t)bn * 16 + (lane & 15)) * FP;
  int c = ct * 16 + (lane & 15);
  bool cok = c < DQ;
  int ccl = cok ? c : (DQ - 1);
  const unsigned short* Bbase = comb + (size_t)bn * FP * DQ;

  floatx8 acc = {0, 0, 0, 0, 0, 0, 0, 0};
#pragma unroll
  for (int k0 = 0; k0 < FP; k0 += 32) {   // 3 chunks, pads zero
    bf16x16 a = ld_frag(Arow, k0 + kh);
    bf16x16 bfrag;
#pragma unroll
    for (int i = 0; i < 16; ++i) {
      int kk = (i < 8) ? (kh + i) : (16 + kh + (i - 8));
      unsigned short u = cok ? Bbase[(size_t)(k0 + kk) * DQ + ccl] : (unsigned short)0;
      bfrag[i] = u16_bf(u);
    }
    acc = wmma_bf16(a, bfrag, acc);
  }
  int rbase = (lane >> 4) * 8;
#pragma unroll
  for (int i = 0; i < 8; ++i) {
    int t = rbase + i;
    if (t < TQ) {
      size_t idx = (((size_t)b * TQ + t) * NQ + n) * DP + c;
      rf[idx] = acc[i];
      rb[idx] = f2bf(acc[i]);
    }
  }
}

// gated fusion: z=sigmoid(re1@W1[t]+gcnx@W2[t]+gen_b[t]); out=z*re1+(1-z)*gcnx
__global__ __launch_bounds__(256) void k_fuse(const unsigned short* __restrict__ r1b,
                                              const unsigned short* __restrict__ gxb,
                                              const float* __restrict__ r1f,
                                              const float* __restrict__ gxf,
                                              const unsigned short* __restrict__ w1, // [12][DP][DP] B^T
                                              const unsigned short* __restrict__ w2,
                                              const float* __restrict__ genb,
                                              float* __restrict__ fused) {
  const int CT = 7, RT = BNQ / 16;
  int wid = (blockIdx.x * blockDim.x + threadIdx.x) >> 5;
  if (wid >= TQ * RT * CT) return;
  int ct = wid % CT; int tmp = wid / CT;
  int rt = tmp % RT;
  int t = tmp / RT;
  int row0 = rt * 16;

  int lane = threadIdx.x & 31;
  int kh = (lane >> 4) * 8;
  int rA = row0 + (lane & 15);
  int bA = rA / NQ;
  int nA = rA - bA * NQ;
  size_t rowBase = (size_t)((bA * TQ + t) * NQ + nA) * DP;
  int c = ct * 16 + (lane & 15);
  const unsigned short* B1 = w1 + ((size_t)t * DP + c) * DP;
  const unsigned short* B2 = w2 + ((size_t)t * DP + c) * DP;

  floatx8 a1c = {0, 0, 0, 0, 0, 0, 0, 0};
  floatx8 a2c = {0, 0, 0, 0, 0, 0, 0, 0};
#pragma unroll
  for (int k0 = 0; k0 < DP; k0 += 32) {
    bf16x16 a1 = ld_frag(r1b + rowBase, k0 + kh);
    bf16x16 a2 = ld_frag(gxb + rowBase, k0 + kh);
    bf16x16 b1 = ld_frag(B1, k0 + kh);
    bf16x16 b2 = ld_frag(B2, k0 + kh);
    a1c = wmma_bf16(a1, b1, a1c);
    a2c = wmma_bf16(a2, b2, a2c);
  }
  int rbase = (lane >> 4) * 8;
  if (c < DQ) {
    float gbv = genb[t * DQ + c];
#pragma unroll
    for (int i = 0; i < 8; ++i) {
      int r = row0 + rbase + i;
      int b = r / NQ;
      int n = r - b * NQ;
      size_t idx = (size_t)((b * TQ + t) * NQ + n) * DP + c;
      float s = a1c[i] + a2c[i] + gbv;
      float z = 1.f / (1.f + __expf(-s));
      fused[idx] = z * r1f[idx] + (1.f - z) * gxf[idx];
    }
  }
}

// head: relu(out@d3+b3)@d4 + b4
__global__ __launch_bounds__(256) void k_head(const float* __restrict__ fused,
                                              const float* __restrict__ d3w, const float* __restrict__ d3b,
                                              const float* __restrict__ d4w, const float* __restrict__ d4b,
                                              float* __restrict__ out) {
  int row = blockIdx.x * blockDim.x + threadIdx.x;
  if (row >= NROWS) return;
  float h[10] = {0, 0, 0, 0, 0, 0, 0, 0, 0, 0};
  const float* x = fused + (size_t)row * DP;
  for (int c = 0; c < DQ; ++c) {
    float xv = x[c];
#pragma unroll
    for (int j = 0; j < 10; ++j) h[j] += xv * d3w[c * 10 + j];
  }
  float y = d4b[0];
#pragma unroll
  for (int j = 0; j < 10; ++j) y += fmaxf(h[j] + d3b[j], 0.f) * d4w[j];
  out[row] = y;
}

// ======================= host launcher =======================

static inline size_t al256(size_t x) { return (x + 255) & ~(size_t)255; }
static inline int cdiv(long a, int b) { return (int)((a + b - 1) / b); }

extern "C" void kernel_launch(void* const* d_in, const int* in_sizes, int n_in,
                              void* d_out, int out_size, void* d_ws, size_t ws_size,
                              hipStream_t stream) {
  (void)in_sizes; (void)n_in; (void)out_size; (void)ws_size;

  const float* time_matrix = (const float*)d_in[0];
  const float* feat  = (const float*)d_in[1];
  const float* Pf    = (const float*)d_in[2];
  const float* Pb    = (const float*)d_in[3];
  const float* d1w = (const float*)d_in[4];  const float* d1b = (const float*)d_in[5];
  const float* d2w = (const float*)d_in[6];  const float* d2b = (const float*)d_in[7];
  const float* d3w = (const float*)d_in[8];  const float* d3b = (const float*)d_in[9];
  const float* d4w = (const float*)d_in[10]; const float* d4b = (const float*)d_in[11];
  const float* gw[3] = {(const float*)d_in[12], (const float*)d_in[13], (const float*)d_in[14]};
  const float* cw[11];
  for (int k = 0; k < 11; ++k) cw[k] = (const float*)d_in[15 + k];
  const float* conv_b = (const float*)d_in[26];
  const float* tbw  = (const float*)d_in[27];
  const float* tbb  = (const float*)d_in[28];
  const float* genw1 = (const float*)d_in[29];
  const float* genw2 = (const float*)d_in[30];
  const float* genb  = (const float*)d_in[31];
  const int*   F     = (const int*)d_in[32];
  float* out = (float*)d_out;

  char* ws = (char*)d_ws;
  size_t off = 0;
  auto alloc = [&](size_t bytes) -> void* { void* p = ws + off; off = al256(off + bytes); return p; };

  size_t xbr_sz  = (size_t)NROWS * DP * 2;
  size_t xbt_sz  = (size_t)NBT * DP * NP * 2;
  size_t t1_sz   = (size_t)3 * NROWS * DP * 2;
  size_t comb_sz = (size_t)BNQ * FP * DQ * 2;
  size_t att_sz  = (size_t)BNQ * 16 * FP * 2;
  size_t timep_sz= (size_t)BQ * TQ * MP * 2;
  size_t P3_sz   = (size_t)3 * NQ * NP * 2;
  size_t gW_sz   = (size_t)6 * DP * DP * 2;
  size_t cW_sz   = (size_t)77 * 128 * 128 * 2;
  size_t genT_sz = (size_t)TQ * DP * DP * 2;

  unsigned short* xbr   = (unsigned short*)alloc(xbr_sz);
  unsigned short* xbt   = (unsigned short*)alloc(xbt_sz);
  unsigned short* t1    = (unsigned short*)alloc(t1_sz);
  unsigned short* comb  = (unsigned short*)alloc(comb_sz);
  unsigned short* att   = (unsigned short*)alloc(att_sz);
  float*          gcnxf = (float*)alloc((size_t)NROWS * DP * 4);
  unsigned short* gcnxb = (unsigned short*)alloc((size_t)NROWS * DP * 2);
  float*          re1f  = (float*)alloc((size_t)NROWS * DP * 4);
  unsigned short* re1b  = (unsigned short*)alloc((size_t)NROWS * DP * 2);
  float*          fused = (float*)alloc((size_t)NROWS * DP * 4);
  unsigned short* timep = (unsigned short*)alloc(timep_sz);
  unsigned short* P3    = (unsigned short*)alloc(P3_sz);
  unsigned short* gW    = (unsigned short*)alloc(gW_sz);
  unsigned short* cW    = (unsigned short*)alloc(cW_sz);
  unsigned short* g1T   = (unsigned short*)alloc(genT_sz);
  unsigned short* g2T   = (unsigned short*)alloc(genT_sz);
  float* rowsum = (float*)alloc(NQ * 4);
  float* colsum = (float*)alloc(NQ * 4);
  float* stot   = (float*)alloc(256);

  const int TB = 256;

  // ---- zero pad regions ----
  hipMemsetAsync(xbt, 0, xbt_sz, stream);
  hipMemsetAsync(att, 0, att_sz, stream);
  hipMemsetAsync(timep, 0, timep_sz, stream);
  hipMemsetAsync(P3, 0, P3_sz, stream);
  hipMemsetAsync(gW, 0, gW_sz, stream);
  hipMemsetAsync(cW, 0, cW_sz, stream);
  hipMemsetAsync(g1T, 0, genT_sz, stream);
  hipMemsetAsync(g2T, 0, genT_sz, stream);
  k_pad_comb<<<cdiv((long)BNQ * 18 * DQ, TB), TB, 0, stream>>>(comb);

  // ---- conversions (into transposed/padded layouts) ----
  k_cvt_time<<<cdiv(BQ * TQ * MQ, TB), TB, 0, stream>>>(time_matrix, timep);
  k_cvt_pmat<<<cdiv(NQ * NQ, TB), TB, 0, stream>>>(Pf, P3);
  k_cvt_pmat<<<cdiv(NQ * NQ, TB), TB, 0, stream>>>(Pb, P3 + (size_t)NQ * NP);
  for (int p = 0; p < 3; ++p)
    k_cvt_sq<<<cdiv(2 * DQ * DQ, TB), TB, 0, stream>>>(gw[p], gW + (size_t)p * 2 * DP * DP, 2);
  static const int wtOffH[11] = {0, 2*128*128, 5*128*128, 9*128*128, 14*128*128,
                                 20*128*128, 27*128*128, 35*128*128, 44*128*128,
                                 54*128*128, 65*128*128};
  for (int k = 0; k < 11; ++k)
    k_cvt_conv<<<cdiv((k + 2) * DQ * DQ, TB), TB, 0, stream>>>(cw[k], cW + wtOffH[k], k + 2);
  k_cvt_sq<<<cdiv(TQ * DQ * DQ, TB), TB, 0, stream>>>(genw1, g1T, TQ);
  k_cvt_sq<<<cdiv(TQ * DQ * DQ, TB), TB, 0, stream>>>(genw2, g2T, TQ);

  // ---- PMI matrix ----
  k_fsum<<<cdiv(NQ, TB), TB, 0, stream>>>(F, rowsum, colsum);
  k_total<<<1, 32, 0, stream>>>(rowsum, stot);
  k_pmi<<<cdiv(NQ * NQ, TB), TB, 0, stream>>>(F, rowsum, colsum, stot,
                                              P3 + (size_t)2 * NQ * NP);

  // ---- embedding MLP (seeds xb_row, xb_t, comb rows 0..11) ----
  k_xb<<<cdiv(NROWS, TB), TB, 0, stream>>>(feat, d1w, d1b, d2w, d2b, xbr, xbt, comb);

  // ---- graph propagation (16x64 register blocking) ----
  { long w = (long)3 * NBT * (NQ / 16) * 2;
    k_gcn_t1<<<cdiv(w * 32, TB), TB, 0, stream>>>(P3, xbt, t1); }
  // ---- GCN projections ----
  { long w = (long)(NROWS / 16) * 8;
    k_gcnx<<<cdiv(w * 32, TB), TB, 0, stream>>>(xbr, t1, gW, gcnxf, gcnxb); }
  // ---- temporal convs (16x64 register blocking) ----
  { long w = (long)26400 * 2;
    k_conv<<<cdiv(w * 32, TB), TB, 0, stream>>>(xbr, cW, conv_b, comb); }
  // ---- temporal-block attention (streams tb_w once) ----
  { long w = (long)TQ * NQ * 5;
    k_att<<<cdiv(w * 32, TB), TB, 0, stream>>>(timep, tbw, tbb, att); }
  // ---- re1 ----
  { long w = (long)BQ * NQ * 8;
    k_re1<<<cdiv(w * 32, TB), TB, 0, stream>>>(att, comb, re1f, re1b); }
  // ---- gated fusion ----
  { long w = (long)TQ * (BNQ / 16) * 7;
    k_fuse<<<cdiv(w * 32, TB), TB, 0, stream>>>(re1b, gcnxb, re1f, gcnxf,
                                                g1T, g2T, genb, fused); }
  // ---- output head ----
  k_head<<<cdiv(NROWS, TB), TB, 0, stream>>>(fused, d3w, d3b, d4w, d4b, out);
}
